// TransformerBlock_33586644255000
// MI455X (gfx1250) — matrix operand order repaired and checked by reading, not compiled, not run on hardware
//
#include <hip/hip_runtime.h>
#include <math.h>

// ---------------- problem constants ----------------
constexpr int BB   = 2;
constexpr int SS   = 2048;
constexpr int DD   = 1024;
constexpr int NH   = 16;
constexpr int HDIM = 64;          // DD / NH
constexpr int FF   = 4096;        // 4*DD
constexpr int ROWS = BB * SS;     // 4096

typedef __bf16 bf16;
typedef __attribute__((ext_vector_type(16))) __bf16 v16bf;
typedef __attribute__((ext_vector_type(8)))  float  v8f;
typedef int v4i_vs __attribute__((vector_size(16)));

// ---------------- async global->LDS copy (CDNA5), guarded ----------------
#if defined(__has_builtin)
#if __has_builtin(__builtin_amdgcn_global_load_async_to_lds_b128)
#define HAVE_ASYNC_LDS 1
#endif
#endif

__device__ __forceinline__ void cp16(const bf16* g, bf16* l) {
#ifdef HAVE_ASYNC_LDS
  __builtin_amdgcn_global_load_async_to_lds_b128(
      (__attribute__((address_space(1))) v4i_vs*)(uintptr_t)g,
      (__attribute__((address_space(3))) v4i_vs*)(uintptr_t)l, 0, 0);
#else
  *(uint4*)l = *(const uint4*)g;
#endif
}
__device__ __forceinline__ void cp_wait() {
#ifdef HAVE_ASYNC_LDS
#if __has_builtin(__builtin_amdgcn_s_wait_asynccnt)
  __builtin_amdgcn_s_wait_asynccnt(0);
#endif
#endif
}

// ---------------- helpers ----------------
__device__ __forceinline__ float red_max16(float v) {
#pragma unroll
  for (int m = 8; m >= 1; m >>= 1) v = fmaxf(v, __shfl_xor(v, m, 32));
  return v;
}
__device__ __forceinline__ float red_sum16(float v) {
#pragma unroll
  for (int m = 8; m >= 1; m >>= 1) v += __shfl_xor(v, m, 32);
  return v;
}

// ---------------- fp32 -> bf16 convert ----------------
__global__ __launch_bounds__(256) void cvt_bf16_kernel(const float* __restrict__ src,
                                                       bf16* __restrict__ dst, int n) {
  int i = blockIdx.x * 256 + threadIdx.x;
  if (i < n) dst[i] = (bf16)src[i];
}

// ---------------- LayerNorm: fp32 in, bf16 out ----------------
__global__ __launch_bounds__(256) void layernorm_kernel(const float* __restrict__ x,
                                                        const float* __restrict__ g,
                                                        const float* __restrict__ be,
                                                        bf16* __restrict__ out) {
  const int row  = blockIdx.x;
  const int tid  = threadIdx.x;
  const int wave = tid >> 5;
  const int lane = tid & 31;
  const float* xr = x + (size_t)row * DD;

  float s = 0.f, s2 = 0.f;
  for (int i = tid; i < DD; i += 256) {
    float v = xr[i];
    s += v; s2 += v * v;
  }
#pragma unroll
  for (int m = 16; m >= 1; m >>= 1) {
    s  += __shfl_xor(s,  m, 32);
    s2 += __shfl_xor(s2, m, 32);
  }
  __shared__ float red[16];
  if (lane == 0) { red[wave] = s; red[8 + wave] = s2; }
  __syncthreads();
  float ts = 0.f, ts2 = 0.f;
#pragma unroll
  for (int w = 0; w < 8; ++w) { ts += red[w]; ts2 += red[8 + w]; }
  const float mu   = ts * (1.0f / DD);
  const float var  = ts2 * (1.0f / DD) - mu * mu;
  const float rstd = rsqrtf(var + 1e-5f);

  bf16* orow = out + (size_t)row * DD;
  for (int i = tid; i < DD; i += 256) {
    float v = (xr[i] - mu) * rstd * g[i] + be[i];
    orow[i] = (bf16)v;
  }
}

// ---------------- WMMA bf16 GEMM (double-buffered, B transposed in LDS) ----------------
// C[M,N] = A[M,K] * B[K,N] (+bias) (gelu?) (+resid) -> Cf (fp32) and/or Cb (bf16)
// Block: 128x128 tile, 256 threads (8 waves). Wave (wm 0..3, wn 0..1) owns 32x64.
#define GBM 128
#define GBN 128
#define GBK 32

__global__ __launch_bounds__(256) void gemm_bf16_kernel(
    const bf16* __restrict__ A, const bf16* __restrict__ Bm,
    const float* __restrict__ bias, const float* __restrict__ resid,
    float* __restrict__ Cf, bf16* __restrict__ Cb,
    int M, int N, int K, int do_gelu) {
  __shared__ __align__(16) bf16 As[2][GBM][GBK];        // 2 x 8 KB
  __shared__ __align__(16) bf16 BsT[2][GBN][GBK + 8];   // 2 x 10 KB (transposed, padded)

  const int tid  = threadIdx.x;
  const int wave = tid >> 5;
  const int lane = tid & 31;
  const int wm   = wave >> 1;     // 0..3
  const int wn   = wave & 1;      // 0..1
  const int h    = lane >> 4;     // half-wave id
  const int l16  = lane & 15;
  const int m0   = blockIdx.y * GBM;
  const int n0   = blockIdx.x * GBN;

  v8f acc[2][4];
#pragma unroll
  for (int mt = 0; mt < 2; ++mt)
#pragma unroll
    for (int nt = 0; nt < 4; ++nt)
#pragma unroll
      for (int i = 0; i < 8; ++i) acc[mt][nt][i] = 0.f;

  // tile-load index precompute
  const int ar = tid >> 1;            // A row within tile (2 threads / row)
  const int ac = (tid & 1) * 16;      // A col within tile
  const int br = tid >> 3;            // B row within tile (8 threads / row)
  const int bc = (tid & 7) * 16;      // B col within tile

  auto stage = [&](int buf, int k0) {
    // A tile 128x32: contiguous 16B chunks -> async direct-to-LDS when available
    const bf16* ga = A + (size_t)(m0 + ar) * K + (k0 + ac);
    cp16(ga,     &As[buf][ar][ac]);
    cp16(ga + 8, &As[buf][ar][ac + 8]);
    // B tile 32x128: load contiguous row chunk, scatter-transpose into BsT
    const bf16* gb = Bm + (size_t)(k0 + br) * N + (n0 + bc);
    bf16 tmp[16];
    *(uint4*)&tmp[0] = *(const uint4*)gb;
    *(uint4*)&tmp[8] = *(const uint4*)(gb + 8);
#pragma unroll
    for (int i = 0; i < 16; ++i) BsT[buf][bc + i][br] = tmp[i];
    if (k0 + 2 * GBK < K) {  // prefetch the tile after next
      __builtin_prefetch(A + (size_t)(m0 + ar) * K + (k0 + 2 * GBK + ac), 0, 3);
      __builtin_prefetch(Bm + (size_t)(k0 + 2 * GBK + br) * N + (n0 + bc), 0, 3);
    }
  };

  const int KT = K / GBK;
  stage(0, 0);
  cp_wait();
  __syncthreads();

  for (int kt = 0; kt < KT; ++kt) {
    const int cur = kt & 1;
    if (kt + 1 < KT) stage(cur ^ 1, (kt + 1) * GBK);

    // ---- fragments + WMMA on current buffer ----
    v16bf afr[2];
#pragma unroll
    for (int mt = 0; mt < 2; ++mt) {
      const bf16* ap = &As[cur][wm * 32 + mt * 16 + l16][0];
#pragma unroll
      for (int e = 0; e < 16; ++e)
        afr[mt][e] = ap[h * 16 + e];                      // contiguous 32B per lane
    }
#pragma unroll
    for (int nt = 0; nt < 4; ++nt) {
      v16bf bfr;
      const bf16* bp = &BsT[cur][wn * 64 + nt * 16 + l16][0];
#pragma unroll
      for (int e = 0; e < 16; ++e)
        bfr[e] = bp[h * 16 + e];                          // contiguous 32B per lane
#pragma unroll
      for (int mt = 0; mt < 2; ++mt)
        acc[mt][nt] = __builtin_amdgcn_wmma_f32_16x16x32_bf16(
            false, afr[mt], false, bfr, (short)0, acc[mt][nt], false, false);
    }
    cp_wait();
    __syncthreads();
  }

  // ---- epilogue: bias / gelu / residual / store ----
#pragma unroll
  for (int mt = 0; mt < 2; ++mt) {
#pragma unroll
    for (int nt = 0; nt < 4; ++nt) {
      const int col = n0 + wn * 64 + nt * 16 + l16;
      const float bv = bias ? bias[col] : 0.f;
#pragma unroll
      for (int r = 0; r < 8; ++r) {
        const int row = m0 + wm * 32 + mt * 16 + r + 8 * h;
        float v = acc[mt][nt][r] + bv;
        if (do_gelu) v = 0.5f * v * (1.0f + erff(v * 0.70710678118f));
        if (resid) v += resid[(size_t)row * N + col];
        if (Cf) Cf[(size_t)row * N + col] = v;
        if (Cb) Cb[(size_t)row * N + col] = (bf16)v;
      }
    }
  }
}

// ---------------- causal flash attention (bf16 WMMA, fp32 softmax) ----------------
// grid: (SS/128, NH, BB); 8 waves, each wave owns 16 query rows.
#define FBQ 128
#define FBK 32
#define NEG_BIG (-3.0e38f)

__global__ __launch_bounds__(256) void attn_kernel(const bf16* __restrict__ Q,
                                                   const bf16* __restrict__ Km,
                                                   const bf16* __restrict__ Vm,
                                                   bf16* __restrict__ ctx) {
  __shared__ __align__(16) bf16 Qs[FBQ][HDIM + 8];      // 18 KB
  __shared__ __align__(16) bf16 Ks[FBK][HDIM + 8];      // [key][d]
  __shared__ __align__(16) bf16 VsT[HDIM][FBK + 8];     // [d][key]  (transposed)
  __shared__ __align__(16) bf16 Ps[8][16][FBK + 8];     // per-wave P scratch

  const int tid  = threadIdx.x;
  const int wave = tid >> 5;
  const int lane = tid & 31;
  const int h    = lane >> 4;
  const int l16  = lane & 15;
  const int b    = blockIdx.z;
  const int hh   = blockIdx.y;
  const int q0   = blockIdx.x * FBQ;

  // ---- stage Q tile (128x64) into LDS, then build per-wave Q fragments ----
  {
    const int qr = tid >> 1;
    const int qc = (tid & 1) * 32;
    const bf16* qg = Q + ((size_t)(b * SS + q0 + qr)) * DD + hh * HDIM + qc;
    *(uint4*)&Qs[qr][qc]      = *(const uint4*)qg;
    *(uint4*)&Qs[qr][qc + 8]  = *(const uint4*)(qg + 8);
    *(uint4*)&Qs[qr][qc + 16] = *(const uint4*)(qg + 16);
    *(uint4*)&Qs[qr][qc + 24] = *(const uint4*)(qg + 24);
  }
  __syncthreads();

  v16bf qa[2];
  {
    const bf16* qp = &Qs[wave * 16 + l16][0];
#pragma unroll
    for (int kk = 0; kk < 2; ++kk)
#pragma unroll
      for (int e = 0; e < 16; ++e)
        qa[kk][e] = qp[kk * 32 + h * 16 + e];
  }

  float mrow[8], lrow[8];
  v8f o[4];
#pragma unroll
  for (int r = 0; r < 8; ++r) { mrow[r] = NEG_BIG; lrow[r] = 0.f; }
#pragma unroll
  for (int ot = 0; ot < 4; ++ot)
#pragma unroll
    for (int i = 0; i < 8; ++i) o[ot][i] = 0.f;

  const int nkb = (q0 + FBQ) / FBK;   // causal: keys in [0, q0+128)
  const int lr = tid >> 3;            // K/V load row
  const int lc = (tid & 7) * 8;       // K/V load col (8 bf16 = 16B)

  for (int kb = 0; kb < nkb; ++kb) {
    const int kbase = kb * FBK;
    // ---- stage K (row-major, async) and V (transposed, sync) tiles ----
    {
      const bf16* kp = Km + ((size_t)(b * SS + kbase + lr)) * DD + hh * HDIM + lc;
      cp16(kp, &Ks[lr][lc]);
      const bf16* vp = Vm + ((size_t)(b * SS + kbase + lr)) * DD + hh * HDIM + lc;
      bf16 tmp[8];
      *(uint4*)tmp = *(const uint4*)vp;
#pragma unroll
      for (int i = 0; i < 8; ++i) VsT[lc + i][lr] = tmp[i];
    }
    cp_wait();
    __syncthreads();

    // ---- S = Q K^T (16 q-rows x 32 keys), two N-tiles, two K-steps each ----
    v8f sc[2];
#pragma unroll
    for (int nt = 0; nt < 2; ++nt) {
#pragma unroll
      for (int i = 0; i < 8; ++i) sc[nt][i] = 0.f;
#pragma unroll
      for (int kk = 0; kk < 2; ++kk) {
        v16bf bfr;
        const bf16* kpR = &Ks[nt * 16 + l16][kk * 32];
#pragma unroll
        for (int e = 0; e < 16; ++e) bfr[e] = kpR[h * 16 + e];
        sc[nt] = __builtin_amdgcn_wmma_f32_16x16x32_bf16(
            false, qa[kk], false, bfr, (short)0, sc[nt], false, false);
      }
    }

    // ---- online softmax update (per owned row r) ----
    const int kj0 = kbase + l16;
    const int kj1 = kbase + 16 + l16;
#pragma unroll
    for (int r = 0; r < 8; ++r) {
      const int qi = q0 + wave * 16 + 8 * h + r;
      float s0 = (kj0 <= qi) ? sc[0][r] * 0.125f : NEG_BIG;   // 1/sqrt(64)
      float s1 = (kj1 <= qi) ? sc[1][r] * 0.125f : NEG_BIG;
      float m = red_max16(fmaxf(s0, s1));
      float mnew = fmaxf(mrow[r], m);
      float corr = __expf(mrow[r] - mnew);
      float p0 = __expf(s0 - mnew);
      float p1 = __expf(s1 - mnew);
      float rs = red_sum16(p0 + p1);
      mrow[r] = mnew;
      lrow[r] = lrow[r] * corr + rs;
#pragma unroll
      for (int ot = 0; ot < 4; ++ot) o[ot][r] *= corr;
      Ps[wave][8 * h + r][l16]      = (bf16)p0;
      Ps[wave][8 * h + r][16 + l16] = (bf16)p1;
    }

    // ---- O += P V : P re-read as A fragment from per-wave LDS ----
    v16bf pa;
    const bf16* pp = &Ps[wave][l16][0];
#pragma unroll
    for (int e = 0; e < 16; ++e)
      pa[e] = pp[h * 16 + e];
#pragma unroll
    for (int ot = 0; ot < 4; ++ot) {
      v16bf vb;
      const bf16* vpR = &VsT[ot * 16 + l16][0];
#pragma unroll
      for (int e = 0; e < 16; ++e) vb[e] = vpR[h * 16 + e];
      o[ot] = __builtin_amdgcn_wmma_f32_16x16x32_bf16(
          false, pa, false, vb, (short)0, o[ot], false, false);
    }
    __syncthreads();
  }

  // ---- normalize and write ctx (bf16, [ROWS, DD] head-interleaved) ----
#pragma unroll
  for (int r = 0; r < 8; ++r) {
    const int qi = q0 + wave * 16 + 8 * h + r;
    const float inv = 1.0f / lrow[r];
    bf16* cp = ctx + ((size_t)(b * SS + qi)) * DD + hh * HDIM;
#pragma unroll
    for (int ot = 0; ot < 4; ++ot)
      cp[ot * 16 + l16] = (bf16)(o[ot][r] * inv);
  }
}

// ---------------- host-side orchestration ----------------
extern "C" void kernel_launch(void* const* d_in, const int* in_sizes, int n_in,
                              void* d_out, int out_size, void* d_ws, size_t ws_size,
                              hipStream_t stream) {
  const float* x  = (const float*)d_in[0];
  const float* Wq = (const float*)d_in[1];
  const float* Wk = (const float*)d_in[2];
  const float* Wv = (const float*)d_in[3];
  const float* Wo = (const float*)d_in[4];
  const float* bo = (const float*)d_in[5];
  const float* W1 = (const float*)d_in[6];
  const float* b1 = (const float*)d_in[7];
  const float* W2 = (const float*)d_in[8];
  const float* b2 = (const float*)d_in[9];
  const float* g1  = (const float*)d_in[10];
  const float* be1 = (const float*)d_in[11];
  const float* g2  = (const float*)d_in[12];
  const float* be2 = (const float*)d_in[13];
  float* out = (float*)d_out;

  // workspace carve-out (256B aligned)
  char* ws = (char*)d_ws;
  size_t off = 0;
  auto carve = [&](size_t bytes) -> void* {
    void* p = ws + off;
    off += (bytes + 255) & ~(size_t)255;
    return p;
  };
  bf16* Wqb = (bf16*)carve((size_t)DD * DD * 2);
  bf16* Wkb = (bf16*)carve((size_t)DD * DD * 2);
  bf16* Wvb = (bf16*)carve((size_t)DD * DD * 2);
  bf16* Wob = (bf16*)carve((size_t)DD * DD * 2);
  bf16* W1b = (bf16*)carve((size_t)DD * FF * 2);
  bf16* W2b = (bf16*)carve((size_t)FF * DD * 2);
  bf16* ln1 = (bf16*)carve((size_t)ROWS * DD * 2);
  bf16* Qb  = (bf16*)carve((size_t)ROWS * DD * 2);
  bf16* Kb  = (bf16*)carve((size_t)ROWS * DD * 2);
  bf16* Vb  = (bf16*)carve((size_t)ROWS * DD * 2);
  bf16* ctx = (bf16*)carve((size_t)ROWS * DD * 2);
  float* x1 = (float*)carve((size_t)ROWS * DD * 4);
  bf16* ln2 = (bf16*)carve((size_t)ROWS * DD * 2);
  bf16* h1  = (bf16*)carve((size_t)ROWS * FF * 2);

  const int nDD = DD * DD, nDF = DD * FF;
  cvt_bf16_kernel<<<(nDD + 255) / 256, 256, 0, stream>>>(Wq, Wqb, nDD);
  cvt_bf16_kernel<<<(nDD + 255) / 256, 256, 0, stream>>>(Wk, Wkb, nDD);
  cvt_bf16_kernel<<<(nDD + 255) / 256, 256, 0, stream>>>(Wv, Wvb, nDD);
  cvt_bf16_kernel<<<(nDD + 255) / 256, 256, 0, stream>>>(Wo, Wob, nDD);
  cvt_bf16_kernel<<<(nDF + 255) / 256, 256, 0, stream>>>(W1, W1b, nDF);
  cvt_bf16_kernel<<<(nDF + 255) / 256, 256, 0, stream>>>(W2, W2b, nDF);

  // ln1 = LayerNorm(x)
  layernorm_kernel<<<ROWS, 256, 0, stream>>>(x, g1, be1, ln1);

  // Q/K/V projections (no bias)
  dim3 gqkv(DD / GBN, ROWS / GBM);
  gemm_bf16_kernel<<<gqkv, 256, 0, stream>>>(ln1, Wqb, nullptr, nullptr, nullptr, Qb,
                                             ROWS, DD, DD, 0);
  gemm_bf16_kernel<<<gqkv, 256, 0, stream>>>(ln1, Wkb, nullptr, nullptr, nullptr, Kb,
                                             ROWS, DD, DD, 0);
  gemm_bf16_kernel<<<gqkv, 256, 0, stream>>>(ln1, Wvb, nullptr, nullptr, nullptr, Vb,
                                             ROWS, DD, DD, 0);

  // causal flash attention
  dim3 gat(SS / FBQ, NH, BB);
  attn_kernel<<<gat, 256, 0, stream>>>(Qb, Kb, Vb, ctx);

  // x1 = x + ctx @ Wo + bo
  gemm_bf16_kernel<<<gqkv, 256, 0, stream>>>(ctx, Wob, bo, x, x1, nullptr,
                                             ROWS, DD, DD, 0);

  // ln2 = LayerNorm(x1)
  layernorm_kernel<<<ROWS, 256, 0, stream>>>(x1, g2, be2, ln2);

  // h1 = gelu(ln2 @ W1 + b1)
  dim3 gff1(FF / GBN, ROWS / GBM);
  gemm_bf16_kernel<<<gff1, 256, 0, stream>>>(ln2, W1b, b1, nullptr, nullptr, h1,
                                             ROWS, FF, DD, 1);

  // out = x1 + h1 @ W2 + b2
  dim3 gff2(DD / GBN, ROWS / GBM);
  gemm_bf16_kernel<<<gff2, 256, 0, stream>>>(h1, W2b, b2, x1, out, nullptr,
                                             ROWS, DD, FF, 0);
}
